// TransformerBlock_5214090297785
// MI455X (gfx1250) — compile-verified, matrix operand order checked by reading, not run on hardware
//
#include <hip/hip_runtime.h>
#include <hip/hip_bf16.h>
#include <math.h>

// ---------------------------------------------------------------------------
// Problem constants (TransformerBlock): N=4, L=2048, C=512, H=8, D=64, FF=2048
// ---------------------------------------------------------------------------
#define NB   4
#define LL   2048
#define CC   512
#define HH   8
#define DD   64
#define FFN  2048
#define MTOK (NB * LL)   // 8192 token rows

typedef __attribute__((ext_vector_type(16))) __bf16 v16bf;
typedef __attribute__((ext_vector_type(8)))  float  v8f;

union FragBF {
  v16bf v;
  unsigned short s[16];
  unsigned int   u[8];
};

__device__ __forceinline__ unsigned short f2bf(float f) {
  union { float f; unsigned int u; } x; x.f = f;
  unsigned int r = x.u + 0x7FFFu + ((x.u >> 16) & 1u);  // round-to-nearest-even
  return (unsigned short)(r >> 16);
}

__device__ __forceinline__ v8f wmma_bf16(const FragBF& a, const FragBF& b, v8f c) {
  return __builtin_amdgcn_wmma_f32_16x16x32_bf16(false, a.v, false, b.v,
                                                 (short)0, c, false, false);
}

// 16x32 bf16 A/B fragment load.  rowPtr = base + (lane%16)*ld.
// element pair ep -> k = (ep>=4 ? 16 : 0) + half*8 + (ep&3)*2  (lowers to ds_load_b128)
__device__ __forceinline__ void frag_load(FragBF& f, const unsigned short* rowPtr, int half) {
#pragma unroll
  for (int ep = 0; ep < 8; ++ep) {
    int kb = ((ep & 4) << 2) + half * 8 + (ep & 3) * 2;
    f.u[ep] = *(const unsigned int*)(rowPtr + kb);
  }
}

__device__ __forceinline__ v8f vzero8() {
  v8f z;
#pragma unroll
  for (int r = 0; r < 8; ++r) z[r] = 0.0f;
  return z;
}

// ---------------------------------------------------------------------------
// CDNA5 async global->LDS copy (16B per lane), ASYNCcnt-tracked.
// LDS byte address = low 32 bits of the generic pointer.
// ---------------------------------------------------------------------------
__device__ __forceinline__ unsigned lds_lo(const void* p) {
  return (unsigned)(unsigned long long)p;
}

__device__ __forceinline__ void async_copy16(unsigned ldsByteAddr, const void* gsrc) {
  asm volatile("global_load_async_to_lds_b128 %0, %1, off"
               :: "v"(ldsByteAddr), "v"(gsrc)
               : "memory");
}

__device__ __forceinline__ void async_wait0() {
  asm volatile("s_wait_asynccnt 0x0" ::: "memory");
}

// ---------------------------------------------------------------------------
// f32 -> bf16 transposing conversion: src [K][Ncols] -> dst [Ncols][K]
// (one-time per launch; makes every GEMM B-tile a natural row-major copy)
// ---------------------------------------------------------------------------
__global__ __launch_bounds__(256) void cvtT_kernel(const float* __restrict__ src,
                                                   unsigned short* __restrict__ dst,
                                                   int K, int Ncols) {
  const int total = K * Ncols;
  for (int i = blockIdx.x * 256 + threadIdx.x; i < total; i += gridDim.x * 256) {
    const int k = i / Ncols, n = i - k * Ncols;
    dst[(size_t)n * K + k] = f2bf(src[i]);
  }
}

// ---------------------------------------------------------------------------
// LayerNorm over C=512, output bf16.  One block per row, 256 threads x 2 cols.
// ---------------------------------------------------------------------------
__global__ __launch_bounds__(256) void ln_kernel(const float* __restrict__ x,
                                                 const float* __restrict__ g,
                                                 const float* __restrict__ b,
                                                 unsigned short* __restrict__ out) {
  const int row = blockIdx.x;
  const int tid = threadIdx.x;
  const float* xr = x + (size_t)row * CC;
  float2 v = *(const float2*)(xr + tid * 2);
  float s  = v.x + v.y;
  float sq = v.x * v.x + v.y * v.y;
#pragma unroll
  for (int off = 16; off > 0; off >>= 1) {
    s  += __shfl_xor(s,  off);
    sq += __shfl_xor(sq, off);
  }
  __shared__ float ss[8], ssq[8];
  const int wave = tid >> 5;
  if ((tid & 31) == 0) { ss[wave] = s; ssq[wave] = sq; }
  __syncthreads();
  float ts = 0.f, tq = 0.f;
#pragma unroll
  for (int i = 0; i < 8; ++i) { ts += ss[i]; tq += ssq[i]; }
  const float mean = ts * (1.0f / CC);
  const float var  = tq * (1.0f / CC) - mean * mean;
  const float rstd = rsqrtf(var + 1e-5f);
  const int c = tid * 2;
  out[(size_t)row * CC + c + 0] = f2bf((v.x - mean) * rstd * g[c + 0] + b[c + 0]);
  out[(size_t)row * CC + c + 1] = f2bf((v.y - mean) * rstd * g[c + 1] + b[c + 1]);
}

// ---------------------------------------------------------------------------
// Async tile staging.  As: 64x32 @stride40.  Bs: 128x32 @stride40 (from W^T).
// 16B-aligned everywhere (stride 40 elem = 80B; chunks at multiples of 16B).
// ---------------------------------------------------------------------------
__device__ __forceinline__ void stage_a_async(unsigned asBase, const unsigned short* A,
                                              int row0, int lda, int k0, int tid) {
  const int row = tid >> 2;
  const int kg  = (tid & 3) * 8;
  async_copy16(asBase + (unsigned)(row * 40 + kg) * 2,
               A + (size_t)(row0 + row) * lda + k0 + kg);
}

__device__ __forceinline__ void stage_b_async(unsigned bsBase, const unsigned short* BT,
                                              int n0, int ldbt, int k0, int tid) {
#pragma unroll
  for (int p = 0; p < 2; ++p) {
    const int linear = tid + p * 256;
    const int n  = linear >> 2;
    const int kg = (linear & 3) * 8;
    async_copy16(bsBase + (unsigned)(n * 40 + kg) * 2,
                 BT + (size_t)(n0 + n) * ldbt + k0 + kg);
  }
}

// ---------------------------------------------------------------------------
// QKV projection: h[8192x512] @ W{q,k,v}^T + bias.
// q,k stored [N,H,L,D]; v stored TRANSPOSED [N,H,D,L] for natural V^T staging.
// ---------------------------------------------------------------------------
__global__ __launch_bounds__(256) void gemm_qkv_kernel(
    const unsigned short* __restrict__ A,
    const unsigned short* __restrict__ WqT, const unsigned short* __restrict__ WkT,
    const unsigned short* __restrict__ WvT,
    const float* __restrict__ bq, const float* __restrict__ bk, const float* __restrict__ bv,
    unsigned short* __restrict__ qo, unsigned short* __restrict__ ko,
    unsigned short* __restrict__ vo) {
  __shared__ __align__(16) unsigned short As[64 * 40];
  __shared__ __align__(16) unsigned short Bs[128 * 40];

  const unsigned short* BT = (blockIdx.z == 0) ? WqT : (blockIdx.z == 1) ? WkT : WvT;
  const float* bias        = (blockIdx.z == 0) ? bq : (blockIdx.z == 1) ? bk : bv;
  unsigned short* dst      = (blockIdx.z == 0) ? qo : (blockIdx.z == 1) ? ko : vo;

  const int tid = threadIdx.x;
  const int wave = tid >> 5, lane = tid & 31, half = lane >> 4, l15 = lane & 15;
  const int waveM = wave >> 2, waveN = wave & 3;
  const int row0 = blockIdx.y * 64, n0 = blockIdx.x * 128;
  const unsigned asB = lds_lo(As), bsB = lds_lo(Bs);

  v8f c[2][2];
  c[0][0] = vzero8(); c[0][1] = vzero8(); c[1][0] = vzero8(); c[1][1] = vzero8();

  for (int k0 = 0; k0 < CC; k0 += 32) {
    stage_a_async(asB, A, row0, CC, k0, tid);
    stage_b_async(bsB, BT, n0, CC, k0, tid);
    async_wait0();
    __syncthreads();
    FragBF af[2], bf[2];
    frag_load(af[0], As + (waveM * 32 + 0 + l15) * 40, half);
    frag_load(af[1], As + (waveM * 32 + 16 + l15) * 40, half);
    frag_load(bf[0], Bs + (waveN * 32 + 0 + l15) * 40, half);
    frag_load(bf[1], Bs + (waveN * 32 + 16 + l15) * 40, half);
#pragma unroll
    for (int i = 0; i < 2; ++i)
#pragma unroll
      for (int j = 0; j < 2; ++j) c[i][j] = wmma_bf16(af[i], bf[j], c[i][j]);
    __syncthreads();
  }

#pragma unroll
  for (int i = 0; i < 2; ++i)
#pragma unroll
    for (int j = 0; j < 2; ++j)
#pragma unroll
      for (int r = 0; r < 8; ++r) {
        const int m  = row0 + waveM * 32 + i * 16 + r + 8 * half;   // token
        const int nn = n0 + waveN * 32 + j * 16 + l15;              // out channel
        const float val = c[i][j][r] + bias[nn];
        const int nb = m >> 11, l = m & (LL - 1);
        const int hh = nn >> 6, d = nn & 63;
        if (blockIdx.z == 2)
          dst[(((size_t)nb * HH + hh) * DD + d) * LL + l] = f2bf(val);  // V^T
        else
          dst[(((size_t)nb * HH + hh) * LL + l) * DD + d] = f2bf(val);
      }
}

// ---------------------------------------------------------------------------
// Flash attention + residual.  grid=(L/128, H, N), 256 thr, wave owns 16 rows.
// K tile [key][d] and V^T tile [d][key] staged with async b128 (stride 72).
// ---------------------------------------------------------------------------
__global__ __launch_bounds__(256) void attn_kernel(
    const unsigned short* __restrict__ qb, const unsigned short* __restrict__ kb,
    const unsigned short* __restrict__ vt, const float* __restrict__ x,
    float* __restrict__ x1) {
  __shared__ __align__(16) unsigned short Kt[64 * 72];    // [key][d]
  __shared__ __align__(16) unsigned short VtT[64 * 72];   // [d][key]
  __shared__ __align__(16) unsigned short Pt[8][16 * 72]; // per-wave [row][key]

  const int n = blockIdx.z, h = blockIdx.y, rb = blockIdx.x;
  const int tid = threadIdx.x;
  const int wave = tid >> 5, lane = tid & 31, half = lane >> 4, l15 = lane & 15;
  const int qRow0 = rb * 128 + wave * 16;
  const size_t headBase = ((size_t)n * HH + h) * LL;
  const unsigned short* vhead = vt + ((size_t)n * HH + h) * DD * LL;  // [d][l]
  const unsigned ktB = lds_lo(Kt), vtB = lds_lo(VtT);
  const float scale = 0.125f;  // 1/sqrt(64)

  FragBF qa[2];
  {
    const unsigned short* qp = qb + (headBase + qRow0 + l15) * DD;
    frag_load(qa[0], qp, half);
    frag_load(qa[1], qp + 32, half);
  }

  float mstate[8], lstate[8];
  v8f oacc[4];
#pragma unroll
  for (int r = 0; r < 8; ++r) { mstate[r] = -3.0e38f; lstate[r] = 0.0f; }
#pragma unroll
  for (int dt = 0; dt < 4; ++dt) oacc[dt] = vzero8();

  for (int j0 = 0; j0 < LL; j0 += 64) {
    // ---- async stage K [key][d] and V^T [d][key] ----
#pragma unroll
    for (int p = 0; p < 2; ++p) {
      const int linear = tid + p * 256;
      const int r  = linear >> 3;        // key (K) / d-row (V^T)
      const int cg = (linear & 7) * 8;   // 8-elem chunk
      async_copy16(ktB + (unsigned)(r * 72 + cg) * 2,
                   kb + (headBase + j0 + r) * DD + cg);
      async_copy16(vtB + (unsigned)(r * 72 + cg) * 2,
                   vhead + (size_t)r * LL + j0 + cg);
    }
    async_wait0();
    __syncthreads();

    // ---- S = Q K^T for 64 keys (4 x 16-key tiles) ----
    v8f sfrag[4];
#pragma unroll
    for (int kt = 0; kt < 4; ++kt) {
      FragBF b0, b1;
      const unsigned short* kp = Kt + (kt * 16 + l15) * 72;
      frag_load(b0, kp, half);
      frag_load(b1, kp + 32, half);
      v8f s0 = wmma_bf16(qa[0], b0, vzero8());
      sfrag[kt] = wmma_bf16(qa[1], b1, s0);
    }

    // ---- online softmax update ----
#pragma unroll
    for (int r = 0; r < 8; ++r) {
      float mx = -3.0e38f;
#pragma unroll
      for (int kt = 0; kt < 4; ++kt) {
        sfrag[kt][r] = sfrag[kt][r] * scale;
        mx = fmaxf(mx, sfrag[kt][r]);
      }
#pragma unroll
      for (int off = 1; off < 16; off <<= 1) mx = fmaxf(mx, __shfl_xor(mx, off));
      const float mnew  = fmaxf(mstate[r], mx);
      const float alpha = __expf(mstate[r] - mnew);
      float rsum = 0.0f;
      const int m = r + 8 * half;
#pragma unroll
      for (int kt = 0; kt < 4; ++kt) {
        const float pv = __expf(sfrag[kt][r] - mnew);
        rsum += pv;
        Pt[wave][m * 72 + kt * 16 + l15] = f2bf(pv);
      }
#pragma unroll
      for (int off = 1; off < 16; off <<= 1) rsum += __shfl_xor(rsum, off);
      lstate[r] = lstate[r] * alpha + rsum;
      mstate[r] = mnew;
#pragma unroll
      for (int dt = 0; dt < 4; ++dt) oacc[dt][r] = oacc[dt][r] * alpha;
    }

    // ---- O += P V ----
    FragBF pa[2];
    {
      const unsigned short* pw = &Pt[wave][0] + l15 * 72;
      frag_load(pa[0], pw, half);
      frag_load(pa[1], pw + 32, half);
    }
#pragma unroll
    for (int dt = 0; dt < 4; ++dt) {
      FragBF v0, v1;
      const unsigned short* vp = VtT + (dt * 16 + l15) * 72;
      frag_load(v0, vp, half);
      frag_load(v1, vp + 32, half);
      oacc[dt] = wmma_bf16(pa[0], v0, oacc[dt]);
      oacc[dt] = wmma_bf16(pa[1], v1, oacc[dt]);
    }
    __syncthreads();
  }

  // ---- finalize: x1 = x + O / l  (proj-out identity) ----
#pragma unroll
  for (int r = 0; r < 8; ++r) {
    const float inv = 1.0f / lstate[r];
    const int m = r + 8 * half;
    const size_t tok = (size_t)n * LL + qRow0 + m;
#pragma unroll
    for (int dt = 0; dt < 4; ++dt) {
      const int cidx = h * DD + dt * 16 + l15;
      x1[tok * CC + cidx] = oacc[dt][r] * inv + x[tok * CC + cidx];
    }
  }
}

// ---------------------------------------------------------------------------
// GEGLU: h2[8192x512] @ Wp^T (+bp), y = u * gelu_exact(g) -> bf16
// Block computes u cols [n0,n0+128) AND g cols [n0+2048, ...) simultaneously.
// ---------------------------------------------------------------------------
__global__ __launch_bounds__(256) void gemm_geglu_kernel(
    const unsigned short* __restrict__ A, const unsigned short* __restrict__ WpT,
    const float* __restrict__ bp, unsigned short* __restrict__ y) {
  __shared__ __align__(16) unsigned short As[64 * 40];
  __shared__ __align__(16) unsigned short Bu[128 * 40];
  __shared__ __align__(16) unsigned short Bg[128 * 40];

  const int tid = threadIdx.x;
  const int wave = tid >> 5, lane = tid & 31, half = lane >> 4, l15 = lane & 15;
  const int waveM = wave >> 2, waveN = wave & 3;
  const int row0 = blockIdx.y * 64, n0 = blockIdx.x * 128;
  const unsigned asB = lds_lo(As), buB = lds_lo(Bu), bgB = lds_lo(Bg);

  v8f cu[2][2], cg[2][2];
#pragma unroll
  for (int i = 0; i < 2; ++i)
#pragma unroll
    for (int j = 0; j < 2; ++j) { cu[i][j] = vzero8(); cg[i][j] = vzero8(); }

  for (int k0 = 0; k0 < CC; k0 += 32) {
    stage_a_async(asB, A, row0, CC, k0, tid);
    stage_b_async(buB, WpT, n0, CC, k0, tid);
    stage_b_async(bgB, WpT, n0 + FFN, CC, k0, tid);
    async_wait0();
    __syncthreads();
    FragBF af[2], bfu[2], bfg[2];
    frag_load(af[0], As + (waveM * 32 + 0 + l15) * 40, half);
    frag_load(af[1], As + (waveM * 32 + 16 + l15) * 40, half);
    frag_load(bfu[0], Bu + (waveN * 32 + 0 + l15) * 40, half);
    frag_load(bfu[1], Bu + (waveN * 32 + 16 + l15) * 40, half);
    frag_load(bfg[0], Bg + (waveN * 32 + 0 + l15) * 40, half);
    frag_load(bfg[1], Bg + (waveN * 32 + 16 + l15) * 40, half);
#pragma unroll
    for (int i = 0; i < 2; ++i)
#pragma unroll
      for (int j = 0; j < 2; ++j) {
        cu[i][j] = wmma_bf16(af[i], bfu[j], cu[i][j]);
        cg[i][j] = wmma_bf16(af[i], bfg[j], cg[i][j]);
      }
    __syncthreads();
  }

#pragma unroll
  for (int i = 0; i < 2; ++i)
#pragma unroll
    for (int j = 0; j < 2; ++j)
#pragma unroll
      for (int r = 0; r < 8; ++r) {
        const int m  = row0 + waveM * 32 + i * 16 + r + 8 * half;
        const int nn = n0 + waveN * 32 + j * 16 + l15;
        const float u = cu[i][j][r] + bp[nn];
        const float g = cg[i][j][r] + bp[nn + FFN];
        const float gel = 0.5f * g * (1.0f + erff(g * 0.70710678118654752f));
        y[(size_t)m * FFN + nn] = f2bf(u * gel);
      }
}

// ---------------------------------------------------------------------------
// Output proj: y[8192x2048] @ W2^T + b2 + x1 -> f32 out
// ---------------------------------------------------------------------------
__global__ __launch_bounds__(256) void gemm_out_kernel(
    const unsigned short* __restrict__ A, const unsigned short* __restrict__ W2T,
    const float* __restrict__ b2, const float* __restrict__ x1,
    float* __restrict__ out) {
  __shared__ __align__(16) unsigned short As[64 * 40];
  __shared__ __align__(16) unsigned short Bs[128 * 40];

  const int tid = threadIdx.x;
  const int wave = tid >> 5, lane = tid & 31, half = lane >> 4, l15 = lane & 15;
  const int waveM = wave >> 2, waveN = wave & 3;
  const int row0 = blockIdx.y * 64, n0 = blockIdx.x * 128;
  const unsigned asB = lds_lo(As), bsB = lds_lo(Bs);

  v8f c[2][2];
  c[0][0] = vzero8(); c[0][1] = vzero8(); c[1][0] = vzero8(); c[1][1] = vzero8();

  for (int k0 = 0; k0 < FFN; k0 += 32) {
    stage_a_async(asB, A, row0, FFN, k0, tid);
    stage_b_async(bsB, W2T, n0, FFN, k0, tid);
    async_wait0();
    __syncthreads();
    FragBF af[2], bf[2];
    frag_load(af[0], As + (waveM * 32 + 0 + l15) * 40, half);
    frag_load(af[1], As + (waveM * 32 + 16 + l15) * 40, half);
    frag_load(bf[0], Bs + (waveN * 32 + 0 + l15) * 40, half);
    frag_load(bf[1], Bs + (waveN * 32 + 16 + l15) * 40, half);
#pragma unroll
    for (int i = 0; i < 2; ++i)
#pragma unroll
      for (int j = 0; j < 2; ++j) c[i][j] = wmma_bf16(af[i], bf[j], c[i][j]);
    __syncthreads();
  }

#pragma unroll
  for (int i = 0; i < 2; ++i)
#pragma unroll
    for (int j = 0; j < 2; ++j)
#pragma unroll
      for (int r = 0; r < 8; ++r) {
        const int m  = row0 + waveM * 32 + i * 16 + r + 8 * half;
        const int nn = n0 + waveN * 32 + j * 16 + l15;
        out[(size_t)m * CC + nn] = c[i][j][r] + b2[nn] + x1[(size_t)m * CC + nn];
      }
}

// ---------------------------------------------------------------------------
// Host-side orchestration
// ---------------------------------------------------------------------------
extern "C" void kernel_launch(void* const* d_in, const int* in_sizes, int n_in,
                              void* d_out, int out_size, void* d_ws, size_t ws_size,
                              hipStream_t stream) {
  (void)in_sizes; (void)n_in; (void)out_size; (void)ws_size;
  const float* x     = (const float*)d_in[0];
  const float* ln0_g = (const float*)d_in[1];
  const float* ln0_b = (const float*)d_in[2];
  const float* Wq    = (const float*)d_in[3];
  const float* bq    = (const float*)d_in[4];
  const float* Wk    = (const float*)d_in[5];
  const float* bk    = (const float*)d_in[6];
  const float* Wv    = (const float*)d_in[7];
  const float* bv    = (const float*)d_in[8];
  const float* ln1_g = (const float*)d_in[9];
  const float* ln1_b = (const float*)d_in[10];
  const float* Wp    = (const float*)d_in[11];
  const float* bp    = (const float*)d_in[12];
  const float* W2    = (const float*)d_in[13];
  const float* b2    = (const float*)d_in[14];
  float* out = (float*)d_out;

  char* ws = (char*)d_ws;
  unsigned short* wqT = (unsigned short*)(ws + 0);          //  512KB [512][512]
  unsigned short* wkT = (unsigned short*)(ws + 524288);     //  512KB
  unsigned short* wvT = (unsigned short*)(ws + 1048576);    //  512KB
  unsigned short* wpT = (unsigned short*)(ws + 1572864);    //    4MB [4096][512]
  unsigned short* w2T = (unsigned short*)(ws + 5767168);    //    2MB [512][2048]
  unsigned short* h_bf= (unsigned short*)(ws + 7864320);    //    8MB (reused for h2)
  unsigned short* q_bf= (unsigned short*)(ws + 16252928);   //    8MB [N,H,L,D]
  unsigned short* k_bf= (unsigned short*)(ws + 24641536);   //    8MB [N,H,L,D]
  unsigned short* v_bf= (unsigned short*)(ws + 33030144);   //    8MB [N,H,D,L] (transposed)
  float*          x1  = (float*)(ws + 41418752);            //   16MB
  unsigned short* y_bf= (unsigned short*)(ws + 58195968);   //   32MB  (total ~87.5MB)

  // 1) weights -> bf16, transposed
  cvtT_kernel<<<512, 256, 0, stream>>>(Wq, wqT, CC, CC);
  cvtT_kernel<<<512, 256, 0, stream>>>(Wk, wkT, CC, CC);
  cvtT_kernel<<<512, 256, 0, stream>>>(Wv, wvT, CC, CC);
  cvtT_kernel<<<2048, 256, 0, stream>>>(Wp, wpT, CC, 2 * FFN);
  cvtT_kernel<<<2048, 256, 0, stream>>>(W2, w2T, FFN, CC);

  // 2) LN0 -> h (bf16)
  ln_kernel<<<MTOK, 256, 0, stream>>>(x, ln0_g, ln0_b, h_bf);

  // 3) QKV projections (V written transposed)
  gemm_qkv_kernel<<<dim3(CC / 128, MTOK / 64, 3), 256, 0, stream>>>(
      h_bf, wqT, wkT, wvT, bq, bk, bv, q_bf, k_bf, v_bf);

  // 4) flash attention + residual -> x1
  attn_kernel<<<dim3(LL / 128, HH, NB), 256, 0, stream>>>(q_bf, k_bf, v_bf, x, x1);

  // 5) LN1 -> h2 (reuse h buffer)
  ln_kernel<<<MTOK, 256, 0, stream>>>(x1, ln1_g, ln1_b, h_bf);

  // 6) GEGLU -> y (bf16)
  gemm_geglu_kernel<<<dim3(FFN / 128, MTOK / 64), 256, 0, stream>>>(h_bf, wpT, bp, y_bf);

  // 7) down-proj + bias + residual -> out (f32)
  gemm_out_kernel<<<dim3(CC / 128, MTOK / 64), 256, 0, stream>>>(y_bf, w2T, b2, x1, out);
}
